// CausalSelfAttention_59554016526603
// MI455X (gfx1250) — matrix operand order repaired and checked by reading, not compiled, not run on hardware
//
#include <hip/hip_runtime.h>
#include <cstdint>

// ---------------- problem constants ----------------
#define B_   2
#define T_   2048
#define E_   2048
#define NH_  16
#define NKV_ 4
#define HD_  128
#define M_   (B_*T_)          // 4096 rows of x
#define DQ_  (NH_*HD_)        // 2048
#define DKV_ (NKV_*HD_)       // 512
#define MASKV (-10000.0f)

typedef __bf16 v16bf __attribute__((ext_vector_type(16)));
typedef __bf16 v8bf  __attribute__((ext_vector_type(8)));
typedef float  v8f   __attribute__((ext_vector_type(8)));
typedef int    v4i_  __attribute__((ext_vector_type(4)));

// ---------------- async global->LDS staging (gfx1250 ASYNCcnt path) ----------------
#if defined(__has_builtin)
# if __has_builtin(__builtin_amdgcn_global_load_async_to_lds_b128)
#  define USE_ASYNC_LDS 1
# endif
#endif
#ifndef USE_ASYNC_LDS
# define USE_ASYNC_LDS 0
#endif

__device__ __forceinline__ void copy16_to_lds(__bf16* l, const __bf16* g) {
#if USE_ASYNC_LDS
  __builtin_amdgcn_global_load_async_to_lds_b128(
      (__attribute__((address_space(1))) v4i_*)g,
      (__attribute__((address_space(3))) v4i_*)l, 0, 0);
#else
  *(uint4*)l = *(const uint4*)g;
#endif
}

__device__ __forceinline__ void wait_lds_copies() {
#if USE_ASYNC_LDS
# if __has_builtin(__builtin_amdgcn_s_wait_asynccnt)
  __builtin_amdgcn_s_wait_asynccnt(0);
# else
  asm volatile("s_wait_asynccnt 0" ::: "memory");
# endif
#endif
}

// ---------------- WMMA helpers ----------------
__device__ __forceinline__ v8f wmma_bf16(v16bf a, v16bf b, v8f c) {
  // D = A(16x32 bf16) * B(32x16 bf16) + C(16x16 f32)
  return __builtin_amdgcn_wmma_f32_16x16x32_bf16(false, a, false, b, (short)0, c, false, false);
}

// A fragment (16x32, bf16), source row-major with given pitch (elements).
// ISA layout: lanes 0-15 -> row m=lane, v0-3 = K0..7, v4-7 = K16..23
//             lanes 16-31 -> row m=lane-16, v0-3 = K8..15, v4-7 = K24..31
__device__ __forceinline__ v16bf load_frag_a(const __bf16* base, int lane, int pitch) {
  int m = lane & 15, ks = lane >> 4;
  const __bf16* r = base + (size_t)m * pitch + ks * 8;
  v8bf lo = *(const v8bf*)(r);        // K[ks*8 .. ks*8+8)
  v8bf hi = *(const v8bf*)(r + 16);   // K[16+ks*8 .. )
  return __builtin_shufflevector(lo, hi, 0,1,2,3,4,5,6,7,8,9,10,11,12,13,14,15);
}

// B fragment (32x16, bf16), source stored "column-major" = N rows of contiguous K.
// ISA layout: lanes 0-15 -> col n=lane, K0..7, K16..23 ; lanes 16-31 -> col n=lane-16, K8..15, K24..31
__device__ __forceinline__ v16bf load_frag_b(const __bf16* base, int lane, int pitch) {
  int n = lane & 15, ks = lane >> 4;
  const __bf16* r = base + (size_t)n * pitch + ks * 8;
  v8bf lo = *(const v8bf*)(r);
  v8bf hi = *(const v8bf*)(r + 16);
  return __builtin_shufflevector(lo, hi, 0,1,2,3,4,5,6,7,8,9,10,11,12,13,14,15);
}

// ---------------- elementwise conversions ----------------
__global__ void f32_to_bf16_kernel(const float* __restrict__ in, __bf16* __restrict__ out, int n) {
  int i = blockIdx.x * blockDim.x + threadIdx.x;
  if (i < n) out[i] = (__bf16)in[i];
}

// W[K,N] f32 -> WT[N,K] bf16
__global__ void transpose_f32_to_bf16_kernel(const float* __restrict__ w, __bf16* __restrict__ wt,
                                             int K, int N) {
  int i = blockIdx.x * blockDim.x + threadIdx.x;
  if (i < K * N) {
    int k = i / N, n = i - k * N;
    wt[(size_t)n * K + k] = (__bf16)w[i];
  }
}

// v[b,t,hkv,d] bf16 -> vt[b,hkv,d,t] bf16
__global__ void v_transpose_kernel(const __bf16* __restrict__ v, __bf16* __restrict__ vt) {
  int i = blockIdx.x * blockDim.x + threadIdx.x;   // over B*T*NKV*HD
  int d = i & (HD_ - 1);
  int rest = i >> 7;            // /HD_
  int h = rest & (NKV_ - 1);
  int bt = rest >> 2;           // /NKV_
  int t = bt & (T_ - 1);
  int b = bt >> 11;             // /T_
  vt[(((size_t)b * NKV_ + h) * HD_ + d) * T_ + t] = v[i];
}

// ---------------- tiled bf16 GEMM: C[M,N] = A[M,K] * WT[N,K]^T ----------------
// block = 256 threads (8 waves); block tile 256x128; wave tile 32x128; K-step 32.
// 16 WMMAs per wave per K-step with B-fragment reuse across the two A-fragments.
template <bool OUT_BF16>
__global__ __launch_bounds__(256) void gemm_bf16_kernel(const __bf16* __restrict__ A,
                                                        const __bf16* __restrict__ WT,
                                                        void* __restrict__ Cout,
                                                        int Mm, int Nn, int Kk) {
  __shared__ __align__(32) __bf16 As[256 * 32];   // 16 KB
  __shared__ __align__(32) __bf16 Bs[128 * 32];   // 8 KB

  const int tid = threadIdx.x, lane = tid & 31, w = tid >> 5;
  const int n0 = blockIdx.x * 128, m0 = blockIdx.y * 256;
  const int nloc = lane & 15, hi8 = (lane >> 4) * 8;

  v8f acc[2][8];
#pragma unroll
  for (int hh = 0; hh < 2; ++hh)
#pragma unroll
    for (int j = 0; j < 8; ++j) acc[hh][j] = (v8f){0,0,0,0,0,0,0,0};

  for (int kk = 0; kk < Kk; kk += 32) {
    // A tile: 256 rows x 32 K = 1024 16B chunks (4/thread)
#pragma unroll
    for (int i = 0; i < 4; ++i) {
      int c = tid + i * 256;
      int row = c >> 2, off = (c & 3) * 8;
      const __bf16* gA = A + (size_t)(m0 + row) * Kk + kk + off;
      copy16_to_lds(As + row * 32 + off, gA);
      __builtin_prefetch(gA + 32, 0, 0);     // next K-slab -> global_prefetch_b8
    }
    // B tile: 128 rows x 32 K = 512 16B chunks (2/thread)
#pragma unroll
    for (int i = 0; i < 2; ++i) {
      int c = tid + i * 256;
      int row = c >> 2, off = (c & 3) * 8;
      const __bf16* gB = WT + (size_t)(n0 + row) * Kk + kk + off;
      copy16_to_lds(Bs + row * 32 + off, gB);
      __builtin_prefetch(gB + 32, 0, 0);
    }
    wait_lds_copies();
    __syncthreads();

    v16bf a0 = load_frag_a(As + (w * 32) * 32, lane, 32);
    v16bf a1 = load_frag_a(As + (w * 32 + 16) * 32, lane, 32);
#pragma unroll
    for (int j = 0; j < 8; ++j) {
      v16bf b = load_frag_b(Bs + (j * 16) * 32, lane, 32);
      acc[0][j] = wmma_bf16(a0, b, acc[0][j]);
      acc[1][j] = wmma_bf16(a1, b, acc[1][j]);
    }
    __syncthreads();
  }

#pragma unroll
  for (int hh = 0; hh < 2; ++hh) {
#pragma unroll
    for (int j = 0; j < 8; ++j) {
#pragma unroll
      for (int r = 0; r < 8; ++r) {
        size_t row = (size_t)(m0 + w * 32 + hh * 16 + r + hi8);
        size_t col = (size_t)(n0 + j * 16 + nloc);
        if (OUT_BF16) ((__bf16*)Cout)[row * Nn + col] = (__bf16)acc[hh][j][r];
        else          ((float*)Cout)[row * Nn + col]  = acc[hh][j][r];
      }
    }
  }
}

// ---------------- RoPE + RMSNorm, in place on bf16 [b,t,heads,HD] ----------------
// one wave32 per (b,t,head); lane handles rotation pairs (lane, lane+64) and (lane+32, lane+96)
__global__ __launch_bounds__(256) void rope_rms_kernel(__bf16* __restrict__ qk, int heads) {
  int gid = blockIdx.x * blockDim.x + threadIdx.x;
  int wid = gid >> 5, lane = threadIdx.x & 31;
  int h = wid % heads;
  int bt = wid / heads;
  int t = bt & (T_ - 1);
  __bf16* base = qk + ((size_t)bt * heads + h) * HD_;

  float x1a = (float)base[lane],      x2a = (float)base[lane + 64];
  float x1b = (float)base[lane + 32], x2b = (float)base[lane + 96];

  float fa = __powf(100000.0f, -(float)lane / 64.0f);
  float fb = __powf(100000.0f, -(float)(lane + 32) / 64.0f);
  float sa, ca, sb, cb;
  __sincosf((float)t * fa, &sa, &ca);
  __sincosf((float)t * fb, &sb, &cb);

  float y1a =  x1a * ca + x2a * sa, y2a = -x1a * sa + x2a * ca;
  float y1b =  x1b * cb + x2b * sb, y2b = -x1b * sb + x2b * cb;

  float ss = y1a * y1a + y2a * y2a + y1b * y1b + y2b * y2b;
#pragma unroll
  for (int m = 16; m >= 1; m >>= 1) ss += __shfl_xor(ss, m, 32);
  float r = rsqrtf(ss * (1.0f / HD_) + 1e-6f);

  base[lane]      = (__bf16)(y1a * r);
  base[lane + 64] = (__bf16)(y2a * r);
  base[lane + 32] = (__bf16)(y1b * r);
  base[lane + 96] = (__bf16)(y2b * r);
}

// ---------------- flash attention ----------------
// grid: b*NH*(T/128) blocks; 8 waves; wave w owns q rows [q0+16w, q0+16w+16)
__global__ __launch_bounds__(256) void flash_attn_kernel(const __bf16* __restrict__ Q,
                                                         const __bf16* __restrict__ K,
                                                         const __bf16* __restrict__ Vt,
                                                         __bf16* __restrict__ Y) {
  __shared__ __align__(32) __bf16 Ks[64 * 128];      // 64 key rows x 128 d
  __shared__ __align__(32) __bf16 Vs[128 * 64];      // 128 d rows x 64 key cols
  __shared__ __align__(32) __bf16 Ps[8 * 16 * 64];   // per-wave P tile

  const int tid = threadIdx.x, lane = tid & 31, w = tid >> 5;
  const int qt = blockIdx.x & 15;                 // T/128 = 16
  const int h  = (blockIdx.x >> 4) & (NH_ - 1);
  const int b  = blockIdx.x >> 8;                 // / (16*NH)
  const int hkv = h >> 2;                         // GQA group = 4
  const int q0 = qt * 128, qw = q0 + w * 16;
  const int nloc = lane & 15, hi8 = (lane >> 4) * 8;
  const float scale = 0.08838834764831845f;       // 1/sqrt(128)

  // Q fragments straight from global (row-major, pitch NH*HD)
  const __bf16* qbase = Q + ((size_t)(b * T_ + qw) * NH_ + h) * HD_;
  v16bf qf[4];
#pragma unroll
  for (int kk = 0; kk < 4; ++kk) qf[kk] = load_frag_a(qbase + kk * 32, lane, NH_ * HD_);

  v8f o[8];
#pragma unroll
  for (int j = 0; j < 8; ++j) o[j] = (v8f){0,0,0,0,0,0,0,0};
  float mrow[8], lrow[8];
#pragma unroll
  for (int r = 0; r < 8; ++r) { mrow[r] = -3.0e38f; lrow[r] = 0.0f; }

  const int ntiles = (q0 + 128) / 64;
  for (int it = 0; it < ntiles; ++it) {
    const int kb = it * 64;

    // cooperative tile staging (16 KB each) via the async-to-LDS path
#pragma unroll
    for (int c0 = 0; c0 < 4; ++c0) {
      int c = tid + c0 * 256;
      int krow = c >> 4, koff = (c & 15) * 8;
      copy16_to_lds(Ks + krow * 128 + koff,
                    K + ((size_t)(b * T_ + kb + krow) * NKV_ + hkv) * HD_ + koff);
      int d = c >> 3, voff = (c & 7) * 8;
      copy16_to_lds(Vs + d * 64 + voff,
                    Vt + (((size_t)b * NKV_ + hkv) * HD_ + d) * T_ + kb + voff);
    }
    wait_lds_copies();
    __syncthreads();

    if (kb <= qw + 15) {  // wave-uniform: skip fully-masked tiles
      // S = Q * K^T  (16 x 64 per wave)
      v8f s[4];
#pragma unroll
      for (int j = 0; j < 4; ++j) s[j] = (v8f){0,0,0,0,0,0,0,0};
#pragma unroll
      for (int kk = 0; kk < 4; ++kk) {
#pragma unroll
        for (int j = 0; j < 4; ++j) {
          v16bf bk = load_frag_b(Ks + (j * 16) * 128 + kk * 32, lane, 128);
          s[j] = wmma_bf16(qf[kk], bk, s[j]);
        }
      }

      // scale + causal mask (reference: scale first, then -10000)
      const bool needMask = (kb + 63 > qw);
#pragma unroll
      for (int j = 0; j < 4; ++j) {
#pragma unroll
        for (int r = 0; r < 8; ++r) {
          float sv = s[j][r] * scale;
          if (needMask) {
            int kidx = kb + j * 16 + nloc;
            int qidx = qw + r + hi8;
            sv = (kidx <= qidx) ? sv : MASKV;
          }
          s[j][r] = sv;
        }
      }

      // online softmax: row m = r + hi8 lives in VGPR r across a 16-lane half
      float mnew[8];
#pragma unroll
      for (int r = 0; r < 8; ++r) {
        float v = fmaxf(fmaxf(s[0][r], s[1][r]), fmaxf(s[2][r], s[3][r]));
#pragma unroll
        for (int m = 8; m >= 1; m >>= 1) v = fmaxf(v, __shfl_xor(v, m, 16));
        mnew[r] = fmaxf(mrow[r], v);
      }

      __bf16* Pw = Ps + (size_t)w * 16 * 64;
#pragma unroll
      for (int r = 0; r < 8; ++r) {
        float alpha = __expf(mrow[r] - mnew[r]);
        float rs = 0.0f;
#pragma unroll
        for (int j = 0; j < 4; ++j) {
          float pexp = __expf(s[j][r] - mnew[r]);
          s[j][r] = pexp;
          rs += pexp;
        }
#pragma unroll
        for (int m = 8; m >= 1; m >>= 1) rs += __shfl_xor(rs, m, 16);
        lrow[r] = lrow[r] * alpha + rs;
        mrow[r] = mnew[r];
#pragma unroll
        for (int jo = 0; jo < 8; ++jo) o[jo][r] *= alpha;
#pragma unroll
        for (int j = 0; j < 4; ++j)
          Pw[(r + hi8) * 64 + j * 16 + nloc] = (__bf16)s[j][r];
      }
      asm volatile("s_wait_dscnt 0" ::: "memory");  // P store -> P A-fragment RAW

      // O += P * V
#pragma unroll
      for (int kk = 0; kk < 2; ++kk) {
        v16bf pa = load_frag_a(Pw + kk * 32, lane, 64);
#pragma unroll
        for (int jo = 0; jo < 8; ++jo) {
          v16bf bv = load_frag_b(Vs + (jo * 16) * 64 + kk * 32, lane, 64);
          o[jo] = wmma_bf16(pa, bv, o[jo]);
        }
      }
    }
    __syncthreads();
  }

  // normalize + store (bf16, [b,t,h,d] == flattened [b,t,E])
#pragma unroll
  for (int r = 0; r < 8; ++r) {
    float inv = 1.0f / lrow[r];
    int tq = qw + r + hi8;
    __bf16* yb = Y + ((size_t)(b * T_ + tq) * NH_ + h) * HD_;
#pragma unroll
    for (int jo = 0; jo < 8; ++jo) yb[jo * 16 + nloc] = (__bf16)(o[jo][r] * inv);
  }
}

// ---------------- driver ----------------
extern "C" void kernel_launch(void* const* d_in, const int* in_sizes, int n_in,
                              void* d_out, int out_size, void* d_ws, size_t ws_size,
                              hipStream_t stream) {
  const float* x  = (const float*)d_in[0];
  const float* wq = (const float*)d_in[1];
  const float* wk = (const float*)d_in[2];
  const float* wv = (const float*)d_in[3];
  const float* wo = (const float*)d_in[4];
  float* out = (float*)d_out;

  char* p = (char*)d_ws;
  auto carve = [&](size_t bytes) -> void* {
    void* r = (void*)p;
    p += (bytes + 255) & ~(size_t)255;
    return r;
  };
  __bf16* xh  = (__bf16*)carve((size_t)M_ * E_ * 2);
  __bf16* wqT = (__bf16*)carve((size_t)DQ_ * E_ * 2);
  __bf16* wkT = (__bf16*)carve((size_t)DKV_ * E_ * 2);
  __bf16* wvT = (__bf16*)carve((size_t)DKV_ * E_ * 2);
  __bf16* woT = (__bf16*)carve((size_t)E_ * E_ * 2);
  __bf16* qh  = (__bf16*)carve((size_t)M_ * DQ_ * 2);
  __bf16* kh  = (__bf16*)carve((size_t)M_ * DKV_ * 2);
  __bf16* vh  = (__bf16*)carve((size_t)M_ * DKV_ * 2);
  __bf16* vt  = (__bf16*)carve((size_t)M_ * DKV_ * 2);
  __bf16* yh  = (__bf16*)carve((size_t)M_ * DQ_ * 2);

  // 1) downconvert
  f32_to_bf16_kernel<<<(M_ * E_) / 256, 256, 0, stream>>>(x, xh, M_ * E_);
  transpose_f32_to_bf16_kernel<<<(E_ * DQ_) / 256, 256, 0, stream>>>(wq, wqT, E_, DQ_);
  transpose_f32_to_bf16_kernel<<<(E_ * DKV_) / 256, 256, 0, stream>>>(wk, wkT, E_, DKV_);
  transpose_f32_to_bf16_kernel<<<(E_ * DKV_) / 256, 256, 0, stream>>>(wv, wvT, E_, DKV_);
  transpose_f32_to_bf16_kernel<<<(E_ * E_) / 256, 256, 0, stream>>>(wo, woT, E_, E_);

  // 2) QKV projections (bf16 out); block tile 256x128
  gemm_bf16_kernel<true><<<dim3(DQ_ / 128, M_ / 256), 256, 0, stream>>>(xh, wqT, qh, M_, DQ_, E_);
  gemm_bf16_kernel<true><<<dim3(DKV_ / 128, M_ / 256), 256, 0, stream>>>(xh, wkT, kh, M_, DKV_, E_);
  gemm_bf16_kernel<true><<<dim3(DKV_ / 128, M_ / 256), 256, 0, stream>>>(xh, wvT, vh, M_, DKV_, E_);

  // 3) RoPE + RMSNorm on q,k ; transpose v for the P*V B-fragment layout
  rope_rms_kernel<<<(B_ * T_ * NH_) * 32 / 256, 256, 0, stream>>>(qh, NH_);
  rope_rms_kernel<<<(B_ * T_ * NKV_) * 32 / 256, 256, 0, stream>>>(kh, NKV_);
  v_transpose_kernel<<<(B_ * T_ * NKV_ * HD_) / 256, 256, 0, stream>>>(vh, vt);

  // 4) flash attention
  flash_attn_kernel<<<B_ * NH_ * (T_ / 128), 256, 0, stream>>>(qh, kh, vt, yh);

  // 5) output projection (f32 out)
  gemm_bf16_kernel<false><<<dim3(E_ / 128, M_ / 256), 256, 0, stream>>>(yh, woT, out, M_, E_, E_);
}
